// GraphMambaLayer_19550691131580
// MI455X (gfx1250) — matrix operand order repaired and checked by reading, not compiled, not run on hardware
//
#include <hip/hip_runtime.h>
#include <hip/hip_bf16.h>
#include <math.h>

// ---------------- sizes (fixed by the reference) ----------------
#define Dm   1024
#define Lm   2
#define NSm  16
#define Bm   64
#define Km   128
#define TOK  (Bm*Km)        // 8192 tokens
#define Hm   32
#define Wm   32

typedef __attribute__((ext_vector_type(16))) _Float16 v16h;
typedef __attribute__((ext_vector_type(8)))  _Float16 v8h;
typedef __attribute__((ext_vector_type(4)))  _Float16 v4h;
typedef __attribute__((ext_vector_type(8)))  float    v8f;

// ============================================================
// WMMA GEMM:  C[M,N] = epilogue(A16[M,K] @ Wt16[N,K]^T + bias)
// A16 row-major f16, Wt16 pre-transposed f16 ([n][k]).
// BM=128, BN=64, BK=32; 128 threads = 4 waves; each wave owns a
// 32x64 strip = 2x4 v_wmma_f32_16x16x32_f16 tiles per k-step.
// Double-buffered LDS (1 barrier / k-step); batched b128 staging;
// all 6 fragments preloaded before the 8-WMMA chain.
// BOUND=true only for N%64!=0 launches (removes exec-mask guards).
// EPI: 0=none 1=gelu 2=softplus 3=+residual 4=(+residual)*rowscale
// ============================================================
#define BMt 128
#define BNt 64
#define BKt 32
#define AP  40   // halves per As row   (80 B rows, 16B-aligned sublines)
#define BP  40   // halves per Bs column

__device__ __forceinline__ float gelu_exact(float x) {
    return 0.5f * x * (1.0f + erff(x * 0.70710678118654752f));
}
__device__ __forceinline__ float softplus_f(float x) {
    return (x > 20.0f) ? x : log1pf(__expf(x));
}

template<int EPI, bool BOUND>
__global__ __launch_bounds__(128)
void gemm_wmma_f16(const _Float16* __restrict__ A16, int lda,
                   const _Float16* __restrict__ Wt, int ldw,   // ldw == K stride of Wt rows
                   const float* __restrict__ bias,
                   float* __restrict__ C, int ldc,
                   _Float16* __restrict__ C16, int ldc16,
                   const float* __restrict__ residual, int ldres,
                   const float* __restrict__ rowscale,
                   int M, int N, int K)
{
    __shared__ _Float16 As[2][BMt * AP];
    __shared__ _Float16 Bs[2][BNt * BP];

    const int tid   = threadIdx.x;
    const int lane  = tid & 31;
    const int wave  = tid >> 5;
    const int half  = lane >> 4;
    const int l16   = lane & 15;

    const int mBase = blockIdx.y * BMt;
    const int nBase = blockIdx.x * BNt;

    v8f acc[2][4];
    #pragma unroll
    for (int rb = 0; rb < 2; ++rb)
        #pragma unroll
        for (int t = 0; t < 4; ++t) acc[rb][t] = (v8f){};

    v8h ar[4];   // A tile 128x32 halves = 512 x 16B chunks
    v8h br[2];   // W tile  64x32 halves = 256 x 16B chunks

    auto loadTile = [&](int k0) {
        #pragma unroll
        for (int i = 0; i < 4; ++i) {
            int c = i * 128 + tid;
            int r = c >> 2, off = (c & 3) * 8;
            ar[i] = *(const v8h*)&A16[(size_t)(mBase + r) * lda + k0 + off];
        }
        #pragma unroll
        for (int i = 0; i < 2; ++i) {
            int c = i * 128 + tid;
            int n = c >> 2, off = (c & 3) * 8;
            if (!BOUND || (nBase + n) < N)
                br[i] = *(const v8h*)&Wt[(size_t)(nBase + n) * ldw + k0 + off];
            else
                br[i] = (v8h){};
        }
    };
    auto storeTile = [&](int p) {
        #pragma unroll
        for (int i = 0; i < 4; ++i) {
            int c = i * 128 + tid;
            int r = c >> 2, off = (c & 3) * 8;
            *(v8h*)&As[p][r * AP + off] = ar[i];
        }
        #pragma unroll
        for (int i = 0; i < 2; ++i) {
            int c = i * 128 + tid;
            int n = c >> 2, off = (c & 3) * 8;
            *(v8h*)&Bs[p][n * BP + off] = br[i];
        }
    };

    loadTile(0);
    storeTile(0);
    __syncthreads();

    int p = 0;
    for (int k0 = 0; k0 < K; k0 += BKt) {
        const bool hasNext = (k0 + BKt) < K;
        if (hasNext) loadTile(k0 + BKt);

        // ---- preload all fragments for this tile (12x ds_load_b128)
        v16h aF[2], bF[4];
        #pragma unroll
        for (int rb = 0; rb < 2; ++rb) {
            const int arow = wave * 32 + rb * 16 + l16;
            v8h alo = *(const v8h*)&As[p][arow * AP + half * 8];
            v8h ahi = *(const v8h*)&As[p][arow * AP + 16 + half * 8];
            aF[rb] = __builtin_shufflevector(alo, ahi,
                     0,1,2,3,4,5,6,7,8,9,10,11,12,13,14,15);
        }
        #pragma unroll
        for (int t = 0; t < 4; ++t) {
            const int bcol = t * 16 + l16;
            v8h blo = *(const v8h*)&Bs[p][bcol * BP + half * 8];
            v8h bhi = *(const v8h*)&Bs[p][bcol * BP + 16 + half * 8];
            bF[t] = __builtin_shufflevector(blo, bhi,
                    0,1,2,3,4,5,6,7,8,9,10,11,12,13,14,15);
        }
        // ---- 8 back-to-back WMMAs
        #pragma unroll
        for (int t = 0; t < 4; ++t) {
            acc[0][t] = __builtin_amdgcn_wmma_f32_16x16x32_f16(
                false, aF[0], false, bF[t], (short)0, acc[0][t], false, false);
            acc[1][t] = __builtin_amdgcn_wmma_f32_16x16x32_f16(
                false, aF[1], false, bF[t], (short)0, acc[1][t], false, false);
        }

        if (hasNext) {
            storeTile(p ^ 1);
            __syncthreads();
        }
        p ^= 1;
    }

    // ---- epilogue + store (C/D layout: VGPR v -> row v + 8*half)
    #pragma unroll
    for (int rb = 0; rb < 2; ++rb) {
        #pragma unroll
        for (int t = 0; t < 4; ++t) {
            int gcol = nBase + t * 16 + l16;
            if (BOUND && gcol >= N) continue;
            float bs = bias ? bias[gcol] : 0.0f;
            #pragma unroll
            for (int v = 0; v < 8; ++v) {
                int grow = mBase + wave * 32 + rb * 16 + half * 8 + v;
                float x = acc[rb][t][v] + bs;
                if (EPI == 1) x = gelu_exact(x);
                else if (EPI == 2) x = softplus_f(x);
                else if (EPI == 3) x = x + residual[(size_t)grow * ldres + gcol];
                else if (EPI == 4) x = (x + residual[(size_t)grow * ldres + gcol]) * rowscale[grow];
                if (C)   C[(size_t)grow * ldc + gcol] = x;
                if (C16) C16[(size_t)grow * ldc16 + gcol] = (_Float16)x;
            }
        }
    }
}

// ============================================================
// weight convert + transpose: Wt[n][k] (f16) <- W[k][n] (f32)
// ============================================================
__global__ __launch_bounds__(256)
void transpose_f16_kernel(const float* __restrict__ W, _Float16* __restrict__ Wt,
                          int K, int N)
{
    __shared__ float tile[32][33];
    const int tx = threadIdx.x & 31, ty = threadIdx.x >> 5;
    const int n0 = blockIdx.x * 32, k0 = blockIdx.y * 32;
    #pragma unroll
    for (int j = 0; j < 4; ++j) {
        int k = k0 + ty + j * 8, n = n0 + tx;
        tile[ty + j * 8][tx] = (k < K && n < N) ? W[(size_t)k * N + n] : 0.0f;
    }
    __syncthreads();
    #pragma unroll
    for (int j = 0; j < 4; ++j) {
        int n = n0 + ty + j * 8, k = k0 + tx;
        if (n < N && k < K) Wt[(size_t)n * K + k] = (_Float16)tile[tx][ty + j * 8];
    }
}

// ============================================================
// centroids + sinusoidal positional features (B*K rows), f16 out
// ============================================================
__global__ __launch_bounds__(256)
void centroid_enc_kernel(const float* __restrict__ attn,
                         float* __restrict__ centroids,
                         _Float16* __restrict__ enc16)
{
    const int row = blockIdx.x;
    const int tid = threadIdx.x;
    __shared__ float rs[256], rx[256], ry[256];
    float s = 0.f, sx = 0.f, sy = 0.f;
    for (int i = tid; i < Hm * Wm; i += 256) {
        float a = attn[(size_t)row * (Hm * Wm) + i];
        float gx = (float)(i % Wm) * (1.0f / (Wm - 1));
        float gy = (float)(i / Wm) * (1.0f / (Hm - 1));
        s += a; sx += a * gx; sy += a * gy;
    }
    rs[tid] = s; rx[tid] = sx; ry[tid] = sy;
    __syncthreads();
    for (int off = 128; off > 0; off >>= 1) {
        if (tid < off) { rs[tid] += rs[tid + off]; rx[tid] += rx[tid + off]; ry[tid] += ry[tid + off]; }
        __syncthreads();
    }
    float inv = 1.0f / (rs[0] + 1e-8f);
    float cx = rx[0] * inv, cy = ry[0] * inv;
    if (tid == 0) { centroids[row * 2 + 0] = cx; centroids[row * 2 + 1] = cy; }
    int c = tid >> 6, jj = tid & 63, j = jj & 31;
    float feat = (c == 0) ? cx : (c == 1) ? cy : 0.1f;
    float freq = __expf(-(float)j * (1.0f / 32.0f) * 9.210340371976184f); // ln(1e4)
    float ang  = feat * freq * 3.14159f;
    float e = (jj < 32) ? sinf(ang) : cosf(ang);
    enc16[(size_t)row * 256 + tid] = (_Float16)e;
}

// ============================================================
// Hilbert order + stable argsort (rank by counting), per batch
// ============================================================
__global__ __launch_bounds__(128)
void hilbert_kernel(const float* __restrict__ centroids,
                    int* __restrict__ order, int* __restrict__ reverse)
{
    const int b = blockIdx.x, k = threadIdx.x;
    float fx = centroids[(b * Km + k) * 2 + 0] * 15.0f;
    float fy = centroids[(b * Km + k) * 2 + 1] * 15.0f;
    fx = fminf(fmaxf(fx, 0.f), 15.f);
    fy = fminf(fmaxf(fy, 0.f), 15.f);
    int x = (int)fx, y = (int)fy, d = 0;
    for (int s = 8; s > 0; s >>= 1) {
        int rxb = (x & s) ? 1 : 0;
        int ryb = (y & s) ? 1 : 0;
        d += s * s * ((3 * rxb) ^ ryb);
        int x1 = rxb ? (s - 1 - x) : x;
        int y1 = rxb ? (s - 1 - y) : y;
        int xn = (ryb == 0) ? y1 : x;
        int yn = (ryb == 0) ? x1 : y;
        x = xn; y = yn;
    }
    __shared__ int dsh[Km];
    dsh[k] = d;
    __syncthreads();
    int rank = 0;
    for (int j = 0; j < Km; ++j) {
        int dj = dsh[j];
        rank += (dj < d) || (dj == d && j < k);
    }
    order[b * Km + rank] = k;
    reverse[b * Km + k] = rank;
}

// ============================================================
// gather to f16: ordered16[b,t,:] = (f16)slots[b, order[b,t], :]
// ============================================================
__global__ __launch_bounds__(256)
void gather_f16_kernel(const float* __restrict__ src, const int* __restrict__ order,
                       _Float16* __restrict__ dst)
{
    const int row = blockIdx.x;
    const int b = row >> 7;
    const int s = order[row];
    const float4 q = ((const float4*)(src + ((size_t)(b * Km + s)) * Dm))[threadIdx.x];
    v4h h;
    h[0] = (_Float16)q.x; h[1] = (_Float16)q.y;
    h[2] = (_Float16)q.z; h[3] = (_Float16)q.w;
    *(v4h*)&dst[(size_t)row * Dm + threadIdx.x * 4] = h;
}

// ============================================================
// selective scan: per (b,d) channel, 16 states, serial in t.
// dir=0 forward, dir=1 backward. f32 in (u,dt), f16 out (ys).
// ============================================================
__global__ __launch_bounds__(256)
void ssm_scan_kernel(const float* __restrict__ u, const float* __restrict__ dt,
                     const float* __restrict__ Bc, const float* __restrict__ Cc,
                     const float* __restrict__ A_log, const float* __restrict__ Dskip,
                     _Float16* __restrict__ ys16, int dir)
{
    const int b = blockIdx.y;
    const int d = blockIdx.x * 256 + threadIdx.x;
    __shared__ float Bsh[NSm], Csh[NSm];
    float Aneg[NSm];
    #pragma unroll
    for (int n = 0; n < NSm; ++n) Aneg[n] = -__expf(A_log[d * NSm + n]);
    const float Ds = Dskip[d];
    float h[NSm] = {};
    for (int s = 0; s < Km; ++s) {
        int t = dir ? (Km - 1 - s) : s;
        size_t tok = (size_t)(b * Km + t);
        __syncthreads();
        if (threadIdx.x < NSm)        Bsh[threadIdx.x]       = Bc[tok * NSm + threadIdx.x];
        else if (threadIdx.x < 2*NSm) Csh[threadIdx.x - NSm] = Cc[tok * NSm + threadIdx.x - NSm];
        __syncthreads();
        float uu = u[tok * Dm + d];
        float dd = dt[tok * Dm + d];
        float y = 0.f;
        #pragma unroll
        for (int n = 0; n < NSm; ++n) {
            h[n] = __expf(dd * Aneg[n]) * h[n] + dd * Bsh[n] * uu;
            y += h[n] * Csh[n];
        }
        ys16[tok * Dm + d] = (_Float16)(y + Ds * uu);
    }
}

// ============================================================
// LayerNorm with optional gather (un-order), residual, f32/f16 out
// ============================================================
__global__ __launch_bounds__(256)
void ln_kernel(const float* __restrict__ x, const int* __restrict__ gatherIdx,
               const float* __restrict__ residual,
               const float* __restrict__ g, const float* __restrict__ bb,
               float* __restrict__ out, _Float16* __restrict__ out16)
{
    const int row = blockIdx.x;
    const int tid = threadIdx.x;
    const int b = row >> 7;
    const int src = gatherIdx ? (b * Km + gatherIdx[row]) : row;
    __shared__ float red[256];
    float v[4];
    float s = 0.f;
    #pragma unroll
    for (int j = 0; j < 4; ++j) {
        int i = tid + j * 256;
        float t = x[(size_t)src * Dm + i];
        if (residual) t += residual[(size_t)row * Dm + i];
        v[j] = t; s += t;
    }
    red[tid] = s; __syncthreads();
    for (int off = 128; off > 0; off >>= 1) {
        if (tid < off) red[tid] += red[tid + off];
        __syncthreads();
    }
    float mean = red[0] * (1.0f / Dm);
    __syncthreads();
    float sq = 0.f;
    #pragma unroll
    for (int j = 0; j < 4; ++j) { float dfl = v[j] - mean; sq += dfl * dfl; }
    red[tid] = sq; __syncthreads();
    for (int off = 128; off > 0; off >>= 1) {
        if (tid < off) red[tid] += red[tid + off];
        __syncthreads();
    }
    float inv = rsqrtf(red[0] * (1.0f / Dm) + 1e-5f);
    #pragma unroll
    for (int j = 0; j < 4; ++j) {
        int i = tid + j * 256;
        float o = (v[j] - mean) * inv * g[i] + bb[i];
        if (out)   out[(size_t)row * Dm + i] = o;
        if (out16) out16[(size_t)row * Dm + i] = (_Float16)o;
    }
}

__global__ __launch_bounds__(256)
void copy_kernel(const float* __restrict__ src, float* __restrict__ dst, int n)
{
    int i = blockIdx.x * 256 + threadIdx.x;
    if (i < n) dst[i] = src[i];
}

// ============================================================
// host launcher
// ============================================================
template<int EPI>
static inline void launch_gemm_e(const _Float16* A, int lda,
                                 const _Float16* Wt, int ldw,
                                 const float* bias, float* C, int ldc,
                                 _Float16* C16, int ldc16,
                                 const float* res, int ldres, const float* rscale,
                                 int M, int N, int K, hipStream_t stream)
{
    dim3 grid((N + BNt - 1) / BNt, (M + BMt - 1) / BMt);
    dim3 blk(128);
    if (N % BNt == 0)
        gemm_wmma_f16<EPI, false><<<grid, blk, 0, stream>>>(A, lda, Wt, ldw, bias, C, ldc, C16, ldc16, res, ldres, rscale, M, N, K);
    else
        gemm_wmma_f16<EPI, true ><<<grid, blk, 0, stream>>>(A, lda, Wt, ldw, bias, C, ldc, C16, ldc16, res, ldres, rscale, M, N, K);
}

static inline void launch_gemm(int epi, const _Float16* A, int lda,
                               const _Float16* Wt, int ldw,
                               const float* bias, float* C, int ldc,
                               _Float16* C16, int ldc16,
                               const float* res, int ldres, const float* rscale,
                               int M, int N, int K, hipStream_t stream)
{
    switch (epi) {
    case 0: launch_gemm_e<0>(A, lda, Wt, ldw, bias, C, ldc, C16, ldc16, res, ldres, rscale, M, N, K, stream); break;
    case 1: launch_gemm_e<1>(A, lda, Wt, ldw, bias, C, ldc, C16, ldc16, res, ldres, rscale, M, N, K, stream); break;
    case 2: launch_gemm_e<2>(A, lda, Wt, ldw, bias, C, ldc, C16, ldc16, res, ldres, rscale, M, N, K, stream); break;
    case 3: launch_gemm_e<3>(A, lda, Wt, ldw, bias, C, ldc, C16, ldc16, res, ldres, rscale, M, N, K, stream); break;
    default: launch_gemm_e<4>(A, lda, Wt, ldw, bias, C, ldc, C16, ldc16, res, ldres, rscale, M, N, K, stream); break;
    }
}

static inline void launch_transpose(const float* W, _Float16* Wt, int K, int N,
                                    hipStream_t stream)
{
    dim3 grid((N + 31) / 32, (K + 31) / 32);
    transpose_f16_kernel<<<grid, 256, 0, stream>>>(W, Wt, K, N);
}

extern "C" void kernel_launch(void* const* d_in, const int* in_sizes, int n_in,
                              void* d_out, int out_size, void* d_ws, size_t ws_size,
                              hipStream_t stream)
{
    (void)in_sizes; (void)n_in; (void)out_size; (void)ws_size;
    const float* in_slots   = (const float*)d_in[0];
    const float* keep_mask  = (const float*)d_in[1];
    const float* attn_maps  = (const float*)d_in[2];
    const float* pos_W1     = (const float*)d_in[5];
    const float* pos_b1     = (const float*)d_in[6];
    const float* pos_W2     = (const float*)d_in[7];
    const float* pos_b2     = (const float*)d_in[8];
    const float* ssm_W_in   = (const float*)d_in[9];
    const float* ssm_b_in   = (const float*)d_in[10];
    const float* ssm_W_dt   = (const float*)d_in[11];
    const float* ssm_b_dt   = (const float*)d_in[12];
    const float* ssm_W_B    = (const float*)d_in[13];
    const float* ssm_W_C    = (const float*)d_in[14];
    const float* ssm_A_log  = (const float*)d_in[15];
    const float* ssm_D      = (const float*)d_in[16];
    const float* ssm_W_out  = (const float*)d_in[17];
    const float* ssm_b_out  = (const float*)d_in[18];
    const float* merge_W    = (const float*)d_in[19];
    const float* merge_b    = (const float*)d_in[20];
    const float* norm_g     = (const float*)d_in[21];
    const float* norm_b     = (const float*)d_in[22];
    const float* ffn_lng    = (const float*)d_in[23];
    const float* ffn_lnb    = (const float*)d_in[24];
    const float* ffn_W1     = (const float*)d_in[25];
    const float* ffn_b1     = (const float*)d_in[26];
    const float* ffn_W2     = (const float*)d_in[27];
    const float* ffn_b2     = (const float*)d_in[28];

    // ---------------- workspace layout ----------------
    float* f = (float*)d_ws;
    float* centroids = f;                 f += (size_t)TOK * 2;
    int*   ordIdx    = (int*)f;           f += TOK;
    int*   revIdx    = (int*)f;           f += TOK;
    float* slots_c   = f;                 f += (size_t)TOK * Dm;
    float* merged    = f;                 f += (size_t)TOK * Dm;
    float* bufU      = f;                 f += (size_t)TOK * Dm;
    float* bufDT     = f;                 f += (size_t)TOK * Dm;
    float* BcBuf     = f;                 f += (size_t)TOK * NSm;
    float* CcBuf     = f;                 f += (size_t)TOK * NSm;

    _Float16* hb = (_Float16*)f;
    _Float16* enc16     = hb;             hb += (size_t)TOK * 256;
    _Float16* ordered16 = hb;             hb += (size_t)TOK * Dm;      // also ph16 / ffnh16 start
    _Float16* ys16      = hb;             hb += (size_t)TOK * Dm;
    _Float16* fb16      = hb;             hb += (size_t)TOK * 2 * Dm;
    _Float16* h16       = hb;             hb += (size_t)TOK * Dm;
    _Float16* ph16      = ordered16;                 // pos hidden (pre-layer)
    _Float16* ffnh16    = ordered16;                 // 8192x4096 = ordered16+ys16+fb16
    // f16 transposed weights
    _Float16* posW1t = hb;  hb += (size_t)Dm * 256;
    _Float16* posW2t = hb;  hb += (size_t)Dm * Dm;
    _Float16* WinT   = hb;  hb += (size_t)4 * Dm * Dm;
    _Float16* WdtT   = hb;  hb += (size_t)4 * Dm * Dm;
    _Float16* WoutT  = hb;  hb += (size_t)4 * Dm * Dm;
    _Float16* WBT    = hb;  hb += (size_t)4 * NSm * Dm;
    _Float16* WCT    = hb;  hb += (size_t)4 * NSm * Dm;
    _Float16* mergeT = hb;  hb += (size_t)2 * Dm * 2 * Dm;
    _Float16* ffnW1T = hb;  hb += (size_t)2 * 4 * Dm * Dm;
    _Float16* ffnW2T = hb;  /* hb += (size_t)2 * 4 * Dm * Dm; */

    // 0) weight convert + transpose (f32 [K][N] -> f16 [N][K])
    launch_transpose(pos_W1, posW1t, 256, Dm, stream);
    launch_transpose(pos_W2, posW2t, Dm, Dm, stream);
    for (int wi = 0; wi < 4; ++wi) {
        launch_transpose(ssm_W_in  + (size_t)wi * Dm * Dm,  WinT  + (size_t)wi * Dm * Dm,  Dm, Dm, stream);
        launch_transpose(ssm_W_dt  + (size_t)wi * Dm * Dm,  WdtT  + (size_t)wi * Dm * Dm,  Dm, Dm, stream);
        launch_transpose(ssm_W_out + (size_t)wi * Dm * Dm,  WoutT + (size_t)wi * Dm * Dm,  Dm, Dm, stream);
        launch_transpose(ssm_W_B   + (size_t)wi * Dm * NSm, WBT   + (size_t)wi * NSm * Dm, Dm, NSm, stream);
        launch_transpose(ssm_W_C   + (size_t)wi * Dm * NSm, WCT   + (size_t)wi * NSm * Dm, Dm, NSm, stream);
    }
    for (int i = 0; i < Lm; ++i) {
        launch_transpose(merge_W + (size_t)i * 2 * Dm * Dm,  mergeT + (size_t)i * Dm * 2 * Dm,  2 * Dm, Dm, stream);
        launch_transpose(ffn_W1  + (size_t)i * Dm * 4 * Dm,  ffnW1T + (size_t)i * 4 * Dm * Dm,  Dm, 4 * Dm, stream);
        launch_transpose(ffn_W2  + (size_t)i * 4 * Dm * Dm,  ffnW2T + (size_t)i * Dm * 4 * Dm,  4 * Dm, Dm, stream);
    }

    // 1) centroids + positional features (f16)
    centroid_enc_kernel<<<TOK, 256, 0, stream>>>(attn_maps, centroids, enc16);

    // 2) pos MLP: ph16 = gelu(enc@W1+b1); slots_c = slots_in + ph@W2+b2
    launch_gemm(1, enc16, 256, posW1t, 256, pos_b1, nullptr, 0, ph16, Dm,
                nullptr, 0, nullptr, TOK, Dm, 256, stream);
    launch_gemm(3, ph16, Dm, posW2t, Dm, pos_b2, slots_c, Dm, nullptr, 0,
                in_slots, Dm, nullptr, TOK, Dm, Dm, stream);

    // 3) Hilbert order / reverse
    hilbert_kernel<<<Bm, Km, 0, stream>>>(centroids, ordIdx, revIdx);

    // 4) layers
    for (int i = 0; i < Lm; ++i) {
        gather_f16_kernel<<<TOK, 256, 0, stream>>>(slots_c, ordIdx, ordered16);

        for (int dir = 0; dir < 2; ++dir) {
            size_t wi = (size_t)(dir * Lm + i);
            const float* bin  = ssm_b_in  + wi * Dm;
            const float* bdt  = ssm_b_dt  + wi * Dm;
            const float* Alog = ssm_A_log + wi * Dm * NSm;
            const float* Dsk  = ssm_D     + wi * Dm;
            const float* bout = ssm_b_out + wi * Dm;

            launch_gemm(0, ordered16, Dm, WinT + wi * Dm * Dm, Dm, bin,
                        bufU, Dm, nullptr, 0, nullptr, 0, nullptr, TOK, Dm, Dm, stream);
            launch_gemm(2, ordered16, Dm, WdtT + wi * Dm * Dm, Dm, bdt,
                        bufDT, Dm, nullptr, 0, nullptr, 0, nullptr, TOK, Dm, Dm, stream);
            launch_gemm(0, ordered16, Dm, WBT + wi * NSm * Dm, Dm, nullptr,
                        BcBuf, NSm, nullptr, 0, nullptr, 0, nullptr, TOK, NSm, Dm, stream);
            launch_gemm(0, ordered16, Dm, WCT + wi * NSm * Dm, Dm, nullptr,
                        CcBuf, NSm, nullptr, 0, nullptr, 0, nullptr, TOK, NSm, Dm, stream);

            ssm_scan_kernel<<<dim3(Dm / 256, Bm), 256, 0, stream>>>(
                bufU, bufDT, BcBuf, CcBuf, Alog, Dsk, ys16, dir);

            // [f | bwd] concat buffer (f16, row stride 2*Dm)
            launch_gemm(0, ys16, Dm, WoutT + wi * Dm * Dm, Dm, bout,
                        nullptr, 0, fb16 + dir * Dm, 2 * Dm,
                        nullptr, 0, nullptr, TOK, Dm, Dm, stream);
        }

        // merged = [f,bwd] @ merge_W[i] + merge_b[i]  (f32 out)
        launch_gemm(0, fb16, 2 * Dm, mergeT + (size_t)i * Dm * 2 * Dm, 2 * Dm,
                    merge_b + i * Dm, merged, Dm, nullptr, 0,
                    nullptr, 0, nullptr, TOK, Dm, 2 * Dm, stream);

        // slots = LN(residual + merged[reverse])   (f32)
        ln_kernel<<<TOK, 256, 0, stream>>>(merged, revIdx, slots_c,
                                           norm_g + i * Dm, norm_b + i * Dm,
                                           slots_c, nullptr);
        // h16 = LN(slots)   (f16, feeds ffn_W1 GEMM)
        ln_kernel<<<TOK, 256, 0, stream>>>(slots_c, nullptr, nullptr,
                                           ffn_lng + i * Dm, ffn_lnb + i * Dm,
                                           nullptr, h16);
        // FFN: slots = (slots + gelu(h@W1+b1)@W2 + b2) * keep_mask
        launch_gemm(1, h16, Dm, ffnW1T + (size_t)i * 4 * Dm * Dm, Dm,
                    ffn_b1 + i * 4 * Dm, nullptr, 0, ffnh16, 4 * Dm,
                    nullptr, 0, nullptr, TOK, 4 * Dm, Dm, stream);
        launch_gemm(4, ffnh16, 4 * Dm, ffnW2T + (size_t)i * Dm * 4 * Dm, 4 * Dm,
                    ffn_b2 + i * Dm, slots_c, Dm, nullptr, 0,
                    slots_c, Dm, keep_mask, TOK, Dm, 4 * Dm, stream);
    }

    // 5) outputs: (slots, centroids) concatenated flat
    float* outp = (float*)d_out;
    copy_kernel<<<(TOK * Dm + 255) / 256, 256, 0, stream>>>(slots_c, outp, TOK * Dm);
    copy_kernel<<<(TOK * 2 + 255) / 256, 256, 0, stream>>>(centroids, outp + (size_t)TOK * Dm, TOK * 2);
}
